// MultiHeadSelfAttention_47674136985927
// MI455X (gfx1250) — matrix operand order repaired and checked by reading, not compiled, not run on hardware
//
#include <hip/hip_runtime.h>

typedef __attribute__((ext_vector_type(16))) __bf16 v16bf;
typedef __attribute__((ext_vector_type(8)))  __bf16 v8bf;
typedef __attribute__((ext_vector_type(4)))  __bf16 v4bf;
typedef __attribute__((ext_vector_type(8)))  float  v8f;
typedef __attribute__((ext_vector_type(4)))  float  f4;

#define NH   16
#define CDIM 1024
#define NSP  64
#define S1   68      // f32 LDS row stride (floats) for the small phase-2 arrays
#define SB1  136     // bf16 LDS row stride, 128-wide K chunks: 272B rows, dword stride 68 (mod64=4)
#define SB2  1048    // bf16 LDS row stride for kernel-2 O1 panel: dword stride 524 (mod64=12)

// ---- WMMA fragment builders (CDNA5 wave32 layouts, ISA 05_wmma 7.12.2) ----
// A (16x32 bf16): lane L holds row L&15; K halves {kb..kb+7, kb+16..kb+23}, kb=8*(L>>4).
static __device__ inline v16bf load_frag_a_bf(const __bf16* base, int S, int kk, int lane) {
  int row = lane & 15;
  int kb  = kk + ((lane >> 4) << 3);
  const __bf16* p = base + row * S + kb;
  v8bf lo = *(const v8bf*)p;           // 16B ds_load_b128
  v8bf hi = *(const v8bf*)(p + 16);    // 16B ds_load_b128
  return __builtin_shufflevector(lo, hi, 0,1,2,3,4,5,6,7,8,9,10,11,12,13,14,15);
}
// B (32x16 bf16): lane L holds col L&15; K halves {kb..kb+7, kb+16..kb+23}, kb=8*(L>>4).
static __device__ inline v16bf load_frag_b_bf(const __bf16* base, int S, int kk, int lane) {
  int col = lane & 15;
  int ks  = kk + ((lane >> 4) << 3);
  const __bf16* p = base + col * S + ks;
  v8bf lo = *(const v8bf*)p;
  v8bf hi = *(const v8bf*)(p + 16);
  return __builtin_shufflevector(lo, hi, 0,1,2,3,4,5,6,7,8,9,10,11,12,13,14,15);
}
// f32-source versions (used only in the tiny attention GEMMs of kernel 1)
static __device__ inline v16bf load_frag_a_f32(const float* base, int S, int kk, int lane) {
  int row = lane & 15;
  int kb  = kk + ((lane >> 4) << 3);
  const float* p = base + row * S + kb;
  v16bf r;
#pragma unroll
  for (int j = 0; j < 8; ++j) r[j]     = (__bf16)p[j];
#pragma unroll
  for (int j = 0; j < 8; ++j) r[8 + j] = (__bf16)p[16 + j];
  return r;
}
static __device__ inline v16bf load_frag_b_f32(const float* base, int S, int kk, int lane) {
  int col = lane & 15;
  int ks  = kk + ((lane >> 4) << 3);
  const float* p = base + col * S + ks;
  v16bf r;
#pragma unroll
  for (int j = 0; j < 16; ++j) r[j] = (__bf16)p[j < 8 ? j : j + 8];
  return r;
}
static __device__ inline v8f wmma_bf16(v16bf a, v16bf b, v8f c) {
  return __builtin_amdgcn_wmma_f32_16x16x32_bf16(false, a, false, b, (short)0, c, false, false);
}
static __device__ inline v4bf cvt4(f4 v) {
  v4bf r; r[0] = (__bf16)v.x; r[1] = (__bf16)v.y; r[2] = (__bf16)v.z; r[3] = (__bf16)v.w;
  return r;
}

// ============================================================================
// Kernel 1: per (batch, head): Q/K/V GEMMs (K=1024), L2-normalize over spatial,
// attn = softmax(Q^T K / T), O = V attn^T  -> o1 (bf16) workspace.
// Wave->tile map: nt (B fragment) fixed per wave -> B loaded once per kk-step;
// A fragments double-buffered (2 live regs) so loads run 2 tiles ahead of WMMA.
// ============================================================================
__global__ __launch_bounds__(256) void
mhsa_qkv_attn_kernel(const float* __restrict__ x,
                     const float* __restrict__ Wq, const float* __restrict__ bq,
                     const float* __restrict__ Wk, const float* __restrict__ bk,
                     const float* __restrict__ Wv, const float* __restrict__ bv,
                     const float* __restrict__ temperature,
                     __bf16* __restrict__ o1) {
  extern __shared__ float sm[];
  // Phase-1 staging (bf16, 128-wide K chunks): 192*136*2 + 64*136*2 = 69,632 B
  __bf16* Wsb = (__bf16*)sm;               // [3*64][SB1]  W slices, k-contiguous
  __bf16* Xsb = (__bf16*)sm + 3 * 64 * SB1;// [64][SB1]    x chunk, transposed [n][k]
  // Phase-2 aliases (f32): 4 * 64*68*4 = 69,632 B (exactly overlaps phase-1 region)
  float* Qs = sm;                 // [n][d]
  float* Ks = sm + 64 * S1;       // [m][d]
  float* Vs = sm + 2 * 64 * S1;   // [d][m]
  float* As = sm + 3 * 64 * S1;   // [n][m]

  const int b    = blockIdx.x >> 4;
  const int h    = blockIdx.x & (NH - 1);
  const int tid  = threadIdx.x;
  const int lane = tid & 31;
  const int wv   = tid >> 5;
  const int ntw  = wv & 3;        // this wave's fixed B (spatial) tile
  const int halfw = wv >> 2;      // selects 6 of the 12 (mu,mt) strips

  // A-strip LDS base for this wave (6 strips at constant offsets from it)
  const __bf16* abase0 = &Wsb[((halfw * 6) >> 2) * 64 * SB1 + (((halfw * 6) & 3) * 16) * SB1];
  const __bf16* abase[6];
#pragma unroll
  for (int i = 0; i < 6; ++i) {
    int pair = halfw * 6 + i;
    int mu = pair >> 2;
    int mt = pair & 3;
    abase[i] = &Wsb[(mu * 64 + mt * 16) * SB1];
  }
  (void)abase0;

  v8f acc[6] = {};

  for (int kc = 0; kc < CDIM; kc += 128) {
    __syncthreads();
    // stage Wq/Wk/Wv head-slices: 3*64*128 f32 -> bf16, 24 float4 per thread
#pragma unroll
    for (int j = 0; j < 24; ++j) {
      int idx = tid + j * 256;             // float4 index 0..6143
      int mu  = idx >> 11;
      int rem = idx & 2047;
      int m   = rem >> 5;
      int kq  = rem & 31;
      const float* Wsel = (mu == 0) ? Wq : ((mu == 1) ? Wk : Wv);
      f4 v = *(const f4*)(Wsel + (size_t)(h * 64 + m) * CDIM + kc + kq * 4);
      *(v4bf*)&Wsb[(mu * 64 + m) * SB1 + kq * 4] = cvt4(v);   // ds_store_b64
    }
    // stage x chunk transposed into Xsb[n][k]; global reads coalesced over n
#pragma unroll
    for (int j = 0; j < 32; ++j) {
      int idx = tid + j * 256;             // 0..8191
      int k = idx >> 6;
      int n = idx & 63;
      Xsb[n * SB1 + k] =
          (__bf16)x[(size_t)b * (CDIM * NSP) + (size_t)(kc + k) * NSP + n];
    }
    if (kc + 128 < CDIM)
      __builtin_prefetch(x + (size_t)b * (CDIM * NSP) + (size_t)(kc + 128) * NSP + tid, 0, 1);
    __syncthreads();
    // 8 waves x 6 tiles; B hoisted, A double-buffered (af0/af1 both live)
#pragma unroll
    for (int kk = 0; kk < 128; kk += 32) {
      v16bf bf  = load_frag_b_bf(&Xsb[(ntw * 16) * SB1], SB1, kk, lane);
      v16bf af0 = load_frag_a_bf(abase[0], SB1, kk, lane);
      v16bf af1 = load_frag_a_bf(abase[1], SB1, kk, lane);
#pragma unroll
      for (int i = 0; i < 6; i += 2) {
        acc[i] = wmma_bf16(af0, bf, acc[i]);
        if (i + 2 < 6) af0 = load_frag_a_bf(abase[i + 2], SB1, kk, lane);
        acc[i + 1] = wmma_bf16(af1, bf, acc[i + 1]);
        if (i + 3 < 6) af1 = load_frag_a_bf(abase[i + 3], SB1, kk, lane);
      }
    }
  }

  const float tinv = 1.0f / (temperature[0] + 1e-6f);
  __syncthreads();
  // spill Q,K,V (+bias) into aliased f32 LDS in WMMA-friendly layouts.
  // D-tile layout: lane -> N = lane&15 (spatial), M rows = vgpr r + 8*(lane>>4).
#pragma unroll
  for (int i = 0; i < 6; ++i) {
    int pair = halfw * 6 + i;
    int mu = pair >> 2;
    int mt = pair & 3;
    int n     = ntw * 16 + (lane & 15);
    int dbase = mt * 16 + ((lane >> 4) << 3);
    const float* bsel = (mu == 0) ? bq : ((mu == 1) ? bk : bv);
#pragma unroll
    for (int r = 0; r < 8; ++r) {
      int d = dbase + r;
      float v = acc[i][r] + bsel[h * 64 + d];
      if (mu == 0)      Qs[n * S1 + d] = v;   // Q^T for attn-A
      else if (mu == 1) Ks[n * S1 + d] = v;   // K   for attn-B ([m][d])
      else              Vs[d * S1 + n] = v;   // V   for out-A  ([d][m])
    }
  }
  __syncthreads();
  // L2-normalize q,k over spatial axis (per d); fold 1/(T+eps) into Q.
  if (tid < 128) {
    float* P = (tid < 64) ? Qs : Ks;
    int d = tid & 63;
    float s = 0.0f;
    for (int n = 0; n < 64; ++n) { float v = P[n * S1 + d]; s += v * v; }
    float inv = 1.0f / fmaxf(sqrtf(s), 1e-12f);
    if (tid < 64) inv *= tinv;
    for (int n = 0; n < 64; ++n) P[n * S1 + d] *= inv;
  }
  __syncthreads();
  // attn[n][m] = sum_d Q^T[n][d] * K[d][m]   (16 tiles, 2 per wave, K=64)
  v8f at[2] = {};
#pragma unroll
  for (int i = 0; i < 2; ++i) {
    int t = wv * 2 + i;
    int mt = t >> 2, nt = t & 3;
#pragma unroll
    for (int kk = 0; kk < 64; kk += 32) {
      v16bf af = load_frag_a_f32(&Qs[(mt * 16) * S1], S1, kk, lane);
      v16bf bf = load_frag_b_f32(&Ks[(nt * 16) * S1], S1, kk, lane);
      at[i] = wmma_bf16(af, bf, at[i]);
    }
  }
#pragma unroll
  for (int i = 0; i < 2; ++i) {
    int t = wv * 2 + i;
    int mt = t >> 2, nt = t & 3;
    int m = nt * 16 + (lane & 15);
    int nbase = mt * 16 + ((lane >> 4) << 3);
#pragma unroll
    for (int r = 0; r < 8; ++r) As[(nbase + r) * S1 + m] = at[i][r];
  }
  __syncthreads();
  // softmax over m, one row per thread (rows contiguous in LDS)
  if (tid < 64) {
    float* row = &As[tid * S1];
    float mx = -3.4e38f;
    for (int m = 0; m < 64; ++m) mx = fmaxf(mx, row[m]);
    float sum = 0.0f;
    for (int m = 0; m < 64; ++m) { float e = __expf(row[m] - mx); row[m] = e; sum += e; }
    float is = 1.0f / sum;
    for (int m = 0; m < 64; ++m) row[m] *= is;
  }
  __syncthreads();
  // O[d][n] = sum_m V[d][m] * attn[n][m]
  v8f oo[2] = {};
#pragma unroll
  for (int i = 0; i < 2; ++i) {
    int t = wv * 2 + i;
    int mt = t >> 2, nt = t & 3;
#pragma unroll
    for (int kk = 0; kk < 64; kk += 32) {
      v16bf af = load_frag_a_f32(&Vs[(mt * 16) * S1], S1, kk, lane);
      v16bf bf = load_frag_b_f32(&As[(nt * 16) * S1], S1, kk, lane);
      oo[i] = wmma_bf16(af, bf, oo[i]);
    }
  }
  size_t ob = ((size_t)b * CDIM + h * 64) * NSP;
#pragma unroll
  for (int i = 0; i < 2; ++i) {
    int t = wv * 2 + i;
    int mt = t >> 2, nt = t & 3;
    int n = nt * 16 + (lane & 15);
    int dbase = mt * 16 + ((lane >> 4) << 3);
#pragma unroll
    for (int r = 0; r < 8; ++r)
      o1[ob + (size_t)(dbase + r) * NSP + n] = (__bf16)oo[i][r];
  }
}

// ============================================================================
// Kernel 2: per batch item: out = Wp * O1 + bp, residual blend, LayerNorm.
// One block owns the whole item: O1 (bf16) lives in LDS for the whole GEMM.
// Wave = mt strip -> A hoisted; B fragments double-buffered.
// ============================================================================
__global__ __launch_bounds__(256) void
mhsa_proj_ln_kernel(const float* __restrict__ x,
                    const float* __restrict__ Wp, const float* __restrict__ bp,
                    const float* __restrict__ gamma,
                    const float* __restrict__ lnw, const float* __restrict__ lnb,
                    const __bf16* __restrict__ o1,
                    float* __restrict__ out) {
  extern __shared__ float sm[];
  __bf16* O1s = (__bf16*)sm;                          // [64][SB2] bf16, k-contiguous
  __bf16* Ws2 = (__bf16*)sm + 64 * SB2;               // [128][SB1] bf16 Wp chunk
  float*  red = (float*)((__bf16*)sm + 64 * SB2 + 128 * SB1);  // [512]

  const int b = blockIdx.x;
  const int tid = threadIdx.x, lane = tid & 31, wv = tid >> 5;

  // load this item's attention output into LDS, transposed to [n][k]
  for (int idx = tid; idx < CDIM * NSP; idx += 256) {
    int k = idx >> 6, n = idx & 63;
    O1s[n * SB2 + k] = o1[(size_t)b * (CDIM * NSP) + idx];
  }
  const float g = gamma[0];
  float lsum = 0.0f, lsq = 0.0f;
  __syncthreads();

  for (int mc = 0; mc < 8; ++mc) {            // 128 output channels per chunk
    v8f acc[4] = {};
    for (int kc = 0; kc < CDIM; kc += 128) {
      __syncthreads();
#pragma unroll
      for (int j = 0; j < 16; ++j) {          // stage Wp chunk: 128x128 f32 -> bf16
        int idx = tid + j * 256;              // float4 index 0..4095
        int m = idx >> 5, kq = idx & 31;
        f4 v = *(const f4*)&Wp[(size_t)(mc * 128 + m) * CDIM + kc + kq * 4];
        *(v4bf*)&Ws2[m * SB1 + kq * 4] = cvt4(v);
      }
      if (kc + 128 < CDIM)
        __builtin_prefetch(Wp + (size_t)(mc * 128) * CDIM + kc + 128 + tid, 0, 1);
      __syncthreads();
#pragma unroll
      for (int kk = 0; kk < 128; kk += 32) {
        v16bf af  = load_frag_a_bf(&Ws2[(wv * 16) * SB1], SB1, kk, lane);  // shared A
        v16bf bf0 = load_frag_b_bf(&O1s[0] + kc, SB2, kk, lane);
        v16bf bf1 = load_frag_b_bf(&O1s[16 * SB2] + kc, SB2, kk, lane);
#pragma unroll
        for (int i = 0; i < 4; i += 2) {      // wave = mt strip, i = nt
          acc[i] = wmma_bf16(af, bf0, acc[i]);
          if (i + 2 < 4) bf0 = load_frag_b_bf(&O1s[((i + 2) * 16) * SB2] + kc, SB2, kk, lane);
          acc[i + 1] = wmma_bf16(af, bf1, acc[i + 1]);
          if (i + 3 < 4) bf1 = load_frag_b_bf(&O1s[((i + 3) * 16) * SB2] + kc, SB2, kk, lane);
        }
      }
    }
    // epilogue: bias + gamma residual; accumulate LN statistics
#pragma unroll
    for (int i = 0; i < 4; ++i) {
      int n = i * 16 + (lane & 15);
      int dbase = mc * 128 + wv * 16 + ((lane >> 4) << 3);
#pragma unroll
      for (int r = 0; r < 8; ++r) {
        int c = dbase + r;
        size_t gi = (size_t)b * (CDIM * NSP) + (size_t)c * NSP + n;
        float v = acc[i][r] + bp[c];
        v = g * v + (1.0f - g) * x[gi];
        out[gi] = v;
        lsum += v;
        lsq  += v * v;
      }
    }
  }
  // block-wide mean / variance over all 65536 elements of this item
  red[tid] = lsum; red[256 + tid] = lsq;
  __syncthreads();
  for (int s = 128; s > 0; s >>= 1) {
    if (tid < s) { red[tid] += red[tid + s]; red[256 + tid] += red[256 + tid + s]; }
    __syncthreads();
  }
  const float inv_n = 1.0f / 65536.0f;
  float mu  = red[0] * inv_n;
  float var = red[256] * inv_n - mu * mu;
  float rs  = rsqrtf(var + 1e-5f);
  __threadfence();
  __syncthreads();
  // normalize in place (reads are L2-hot: this block just wrote them)
  for (int idx = tid; idx < CDIM * NSP; idx += 256) {
    size_t gi = (size_t)b * (CDIM * NSP) + idx;
    float v = out[gi];
    out[gi] = (v - mu) * rs * lnw[idx] + lnb[idx];
  }
}

extern "C" void kernel_launch(void* const* d_in, const int* in_sizes, int n_in,
                              void* d_out, int out_size, void* d_ws, size_t ws_size,
                              hipStream_t stream) {
  const float* x    = (const float*)d_in[0];
  const float* Wq   = (const float*)d_in[1];
  const float* bq   = (const float*)d_in[2];
  const float* Wk   = (const float*)d_in[3];
  const float* bk   = (const float*)d_in[4];
  const float* Wv   = (const float*)d_in[5];
  const float* bv   = (const float*)d_in[6];
  const float* Wp   = (const float*)d_in[7];
  const float* bp   = (const float*)d_in[8];
  const float* gam  = (const float*)d_in[9];
  const float* temp = (const float*)d_in[10];
  const float* lnw  = (const float*)d_in[11];
  const float* lnb  = (const float*)d_in[12];
  float*   out = (float*)d_out;
  __bf16*  o1  = (__bf16*)d_ws;   // 1024*1024*64 bf16 = 128 MiB

  const size_t sh1 = (size_t)(3 * 64 * SB1 + 64 * SB1) * 2;                    // 69,632 B
  const size_t sh2 = (size_t)(64 * SB2 + 128 * SB1) * 2 + 512 * 4;             // ~171 KiB
  hipFuncSetAttribute((const void*)mhsa_qkv_attn_kernel,
                      hipFuncAttributeMaxDynamicSharedMemorySize, (int)sh1);
  hipFuncSetAttribute((const void*)mhsa_proj_ln_kernel,
                      hipFuncAttributeMaxDynamicSharedMemorySize, (int)sh2);

  mhsa_qkv_attn_kernel<<<1024 * NH, 256, sh1, stream>>>(x, Wq, bq, Wk, bk, Wv, bv, temp, o1);
  mhsa_proj_ln_kernel<<<1024, 256, sh2, stream>>>(x, Wp, bp, gam, lnw, lnb, o1, out);
}
